// ConvAttention_74783970558500
// MI455X (gfx1250) — matrix operand order repaired and checked by reading, not compiled, not run on hardware
//
#include <hip/hip_runtime.h>

// ---------------------------------------------------------------------------
// ConvAttention (Restormer transposed attention) for MI455X / gfx1250.
//
// Pipeline (all bf16 WMMA 16x16x32, f32 accumulate):
//   [0] cvt   : x -> xTp bf16 channel-last, zero-padded [n][130][130][ic]
//               (pad removes ALL boundary branches from the conv hot loop)
//               qkv_w repacked [m][tap][ic] bf16 ; proj_w bf16
//   [1] conv  : 9-tap implicit GEMM, M=576 K=1728 N=131072, branch-free
//               B-fragments = two 16B loads per lane from xTp
//               writes q,k as [n][c192][p] and v transposed [n][p][c192]
//   [2] norm  : 1/max(||q_c||,eps), 1/max(||k_c||,eps) over HW=16384
//   [3] attn  : 32x32 = q . k^T over K=16384 (8 waves split K, LDS reduce),
//               scale by invq*invk*temperature, softmax, store bf16
//   [4] av    : out = attn @ v  (M=32,K=32,N=16384), writes outT [n][p][c192]
//   [5] proj  : 192x192x131072 GEMM + bias -> d_out f32
//
// MI455X reasoning: xTp (52MB) and per-row working sets live in the 192MB L2,
// so the 9x re-gather of shifted positions in the tap-decomposed conv costs
// L2 BW (2KB per 16KFLOP wmma = 8 FLOP/B), not HBM. HBM-critical traffic
// ~= 100MB in + 100MB out (~9us at 23.3 TB/s); conv WMMA is the anchor.
// ---------------------------------------------------------------------------

typedef unsigned short u16;
typedef unsigned int   u32;
typedef __attribute__((ext_vector_type(16))) __bf16 bf16x16;
typedef __attribute__((ext_vector_type(8)))  float  f32x8;
typedef __attribute__((ext_vector_type(4)))  u32    u32x4;

#define HW    16384
#define DIM   192
#define HEADS 6
#define PW    130          // padded width/height (128 + 1 halo each side)
#define PROW  (PW * DIM)   // elements per padded row

union BFrag {
  bf16x16 v;
  u16     u[16];
  u32x4   q[2];
};

__device__ __forceinline__ u16 f32_to_bf16(float f) {
  u32 u = __float_as_uint(f);
  u32 r = u + 0x7FFFu + ((u >> 16) & 1u);   // round-to-nearest-even
  return (u16)(r >> 16);
}
__device__ __forceinline__ float bf16_to_f32(u16 h) {
  return __uint_as_float(((u32)h) << 16);
}
__device__ __forceinline__ f32x8 wmma_bf16(const BFrag& a, const BFrag& b, f32x8 c) {
  return __builtin_amdgcn_wmma_f32_16x16x32_bf16(
      /*neg_a=*/false, a.v, /*neg_b=*/false, b.v,
      /*c_mod=*/(short)0, c, /*reuse_a=*/false, /*reuse_b=*/false);
}
__device__ __forceinline__ u32x4 pack8_bf16(f32x8 v) {
  union { u32x4 q; u16 u[8]; } r;
#pragma unroll
  for (int g = 0; g < 8; ++g) r.u[g] = f32_to_bf16(v[g]);
  return r.q;
}

// ------------------------------ [0] converts -------------------------------
__global__ void zero_u32x4_kernel(u32x4* __restrict__ dst, int n16) {
  int i = blockIdx.x * 256 + threadIdx.x;
  const u32x4 z = {0u, 0u, 0u, 0u};
  if (i < n16) dst[i] = z;
}

__global__ void cvt_f32_bf16_kernel(const float* __restrict__ src,
                                    u16* __restrict__ dst, int n) {
  int i = blockIdx.x * 256 + threadIdx.x;
  if (i < n) dst[i] = f32_to_bf16(src[i]);
}

// x [n][c][128*128] f32 -> xTp [n][130][130][c] bf16 interior at (+1,+1).
// Block handles 64 positions x 192 channels via LDS tile (coalesced both ways)
__global__ void cvt_x_transpose_kernel(const float* __restrict__ x,
                                       u16* __restrict__ xTp) {
  __shared__ u16 tile[64][DIM + 8];
  const size_t p0 = (size_t)blockIdx.x * 64;      // 2048 blocks
  const int n  = (int)(p0 >> 14);
  const int pr = (int)(p0 & (HW - 1));
  const int pl = threadIdx.x & 63;                // coalesced along p
  const int c0 = threadIdx.x >> 6;                // 0..3
  for (int c = c0; c < DIM; c += 4)
    tile[pl][c] = f32_to_bf16(x[((size_t)n * DIM + c) * HW + pr + pl]);
  __syncthreads();
  for (int e = threadIdx.x; e < 64 * DIM; e += 256) {
    const int p = e / DIM, c = e % DIM;           // coalesced along c
    const int pp = pr + p;
    const int y = pp >> 7, xx = pp & 127;
    xTp[((size_t)n * PW + (y + 1)) * PROW + (size_t)(xx + 1) * DIM + c] = tile[p][c];
  }
}

// qkv_w [576][192][3][3] -> wpack [576][tap(9)][ic(192)]  (contiguous K steps)
__global__ void pack_qkvw_kernel(const float* __restrict__ w,
                                 u16* __restrict__ dst) {
  int i = blockIdx.x * 256 + threadIdx.x;
  if (i < 576 * 1728) {
    int m = i / 1728, k = i % 1728;
    int r = k / DIM, ic = k % DIM;
    dst[i] = f32_to_bf16(w[(size_t)m * 1728 + (size_t)ic * 9 + r]);
  }
}

// ------------------------ [1] qkv conv (implicit GEMM) ---------------------
// grid (1024 image-rows, 9 m-tiles), block 256 (8 waves).
// Wave (wid): D-stripe 16M x 64N; block tile 64M x 128N (one full image row).
// Zero-padded input => completely branch-free inner loop.
__global__ void qkv_conv_kernel(const u16* __restrict__ xTp,
                                const u16* __restrict__ wpack,
                                u16* __restrict__ qbf,
                                u16* __restrict__ kbf,
                                u16* __restrict__ vT) {
  const int lane = threadIdx.x & 31;
  const int wid  = threadIdx.x >> 5;
  const int hi   = lane >> 4;
  const int ln   = lane & 15;

  const int prow  = blockIdx.x;       // 0..1023 = (n, y)
  const int n     = prow >> 7;
  const int y     = prow & 127;
  const int mbase = blockIdx.y * 64 + (wid >> 1) * 16;
  const int nbase = (wid & 1) * 64;

  const u16* xTn = xTp + (size_t)n * PW * PROW;
  const u16* wrow = wpack + (size_t)(mbase + ln) * 1728 + hi * 8;

  const f32x8 z = {0.f, 0.f, 0.f, 0.f, 0.f, 0.f, 0.f, 0.f};
  f32x8 acc[4] = {z, z, z, z};

  for (int tap = 0; tap < 9; ++tap) {
    const int dy = tap / 3, dx = tap % 3;         // padded offsets (0..2)
    // base of padded position (y+dy, nbase+ln+dx) in channel-last layout
    const u16* xbase = xTn + ((size_t)(y + dy) * PW + nbase + ln + dx) * DIM + hi * 8;

#pragma unroll 2
    for (int ks = 0; ks < 6; ++ks) {
      const int icb = ks * 32;
      // A fragment: weight row m=mbase+ln, K group {hi*8..+7, 16+hi*8..+7}
      BFrag a;
      const u16* arow = wrow + tap * DIM + icb;
      a.q[0] = *(const u32x4*)(arow);
      a.q[1] = *(const u32x4*)(arow + 16);
      __builtin_prefetch(arow + DIM, 0, 3);  // next tap's weights, keep near

#pragma unroll
      for (int sub = 0; sub < 4; ++sub) {
        // B fragment: ic {hi*8..+7, 16+hi*8..+7} at padded position -> two b128 loads
        const u16* bp = xbase + (size_t)sub * 16 * DIM + icb;
        BFrag b;
        b.q[0] = *(const u32x4*)bp;
        b.q[1] = *(const u32x4*)(bp + 16);
        acc[sub] = wmma_bf16(a, b, acc[sub]);
      }
    }
  }

#pragma unroll
  for (int sub = 0; sub < 4; ++sub) {
    const int p = y * 128 + nbase + sub * 16 + ln;  // position within image
    if (mbase < 192) {
#pragma unroll
      for (int g = 0; g < 8; ++g)
        qbf[((size_t)n * DIM + mbase + g + hi * 8) * HW + p] = f32_to_bf16(acc[sub][g]);
    } else if (mbase < 384) {
#pragma unroll
      for (int g = 0; g < 8; ++g)
        kbf[((size_t)n * DIM + (mbase - 192) + g + hi * 8) * HW + p] = f32_to_bf16(acc[sub][g]);
    } else {
      // v channels are consecutive per lane -> single 16B packed store
      *(u32x4*)(vT + ((size_t)n * HW + p) * DIM + (mbase - 384) + hi * 8) =
          pack8_bf16(acc[sub]);
    }
  }
}

// ------------------------------ [2] norms ----------------------------------
// grid 48 = (n,head); block 256 = 8 waves; 64 rows (32 q + 32 k), 8 per wave.
__global__ void norm_kernel(const u16* __restrict__ qbf,
                            const u16* __restrict__ kbf,
                            float* __restrict__ invq,
                            float* __restrict__ invk) {
  const int nh = blockIdx.x;
  const int n = nh / HEADS, h = nh % HEADS;
  const int wid = threadIdx.x >> 5, lane = threadIdx.x & 31;
  for (int rr = 0; rr < 8; ++rr) {
    const int row = wid * 8 + rr;
    const int c = row & 31;
    const u16* src = (row < 32 ? qbf : kbf) + ((size_t)n * DIM + h * 32 + c) * HW;
    float ss = 0.f;
    for (int i = lane; i < HW; i += 32) {
      float v = bf16_to_f32(src[i]);
      ss += v * v;
    }
#pragma unroll
    for (int o = 16; o > 0; o >>= 1) ss += __shfl_down(ss, o, 32);
    if (lane == 0) {
      float inv = 1.0f / fmaxf(sqrtf(ss), 1e-12f);
      (row < 32 ? invq : invk)[nh * 32 + c] = inv;
    }
  }
}

// ------------------ [3] attention logits + softmax -------------------------
// grid 48; 8 waves each do the 32x32 GEMM over K=2048, LDS reduce, softmax.
__global__ void attn_kernel(const u16* __restrict__ qbf,
                            const u16* __restrict__ kbf,
                            const float* __restrict__ invq,
                            const float* __restrict__ invk,
                            const float* __restrict__ temp,
                            u16* __restrict__ attn_bf) {
  __shared__ float part[8][32][32];
  __shared__ float red[32][33];
  const int nh = blockIdx.x;
  const int n = nh / HEADS, h = nh % HEADS;
  const int wid = threadIdx.x >> 5, lane = threadIdx.x & 31;
  const int hi = lane >> 4, ln = lane & 15;
  const u16* qb = qbf + ((size_t)n * DIM + h * 32) * HW;
  const u16* kb = kbf + ((size_t)n * DIM + h * 32) * HW;

  const f32x8 z = {0.f, 0.f, 0.f, 0.f, 0.f, 0.f, 0.f, 0.f};
  f32x8 acc[2][2] = {{z, z}, {z, z}};
  const int k0 = wid * 2048;

  for (int kk = 0; kk < 2048; kk += 32) {
    const int kbase = k0 + kk;
    BFrag aq[2], bk[2];
#pragma unroll
    for (int t = 0; t < 2; ++t) {
      const u16* ap = qb + (size_t)(t * 16 + ln) * HW + kbase + hi * 8;
      aq[t].q[0] = *(const u32x4*)ap;
      aq[t].q[1] = *(const u32x4*)(ap + 16);
      const u16* bp = kb + (size_t)(t * 16 + ln) * HW + kbase + hi * 8;
      bk[t].q[0] = *(const u32x4*)bp;
      bk[t].q[1] = *(const u32x4*)(bp + 16);
    }
#pragma unroll
    for (int i = 0; i < 2; ++i)
#pragma unroll
      for (int j = 0; j < 2; ++j) acc[i][j] = wmma_bf16(aq[i], bk[j], acc[i][j]);
  }

#pragma unroll
  for (int i = 0; i < 2; ++i)
#pragma unroll
    for (int j = 0; j < 2; ++j)
#pragma unroll
      for (int g = 0; g < 8; ++g)
        part[wid][i * 16 + g + hi * 8][j * 16 + ln] = acc[i][j][g];
  __syncthreads();

  for (int e = threadIdx.x; e < 1024; e += 256) {
    const int c = e >> 5, d = e & 31;
    float s = 0.f;
#pragma unroll
    for (int w = 0; w < 8; ++w) s += part[w][c][d];
    red[c][d] = s * invq[nh * 32 + c] * invk[nh * 32 + d] * temp[h];
  }
  __syncthreads();

  if (threadIdx.x < 32) {
    const int c = threadIdx.x;
    float mx = -1e30f;
    for (int d = 0; d < 32; ++d) mx = fmaxf(mx, red[c][d]);
    float sum = 0.f;
    for (int d = 0; d < 32; ++d) sum += __expf(red[c][d] - mx);
    const float rs = 1.f / sum;
    for (int d = 0; d < 32; ++d)
      attn_bf[(size_t)nh * 1024 + c * 32 + d] = f32_to_bf16(__expf(red[c][d] - mx) * rs);
  }
}

// ----------------------------- [4] attn @ v --------------------------------
// grid (128 p-tiles, 48); 8 waves each do M=32,K=32,N=16 -> outT [n][p][c192]
__global__ void av_kernel(const u16* __restrict__ attn_bf,
                          const u16* __restrict__ vT,
                          u16* __restrict__ outT) {
  const int nh = blockIdx.y;
  const int n = nh / HEADS, h = nh % HEADS;
  const int wid = threadIdx.x >> 5, lane = threadIdx.x & 31;
  const int hi = lane >> 4, ln = lane & 15;
  const int p = blockIdx.x * 128 + wid * 16 + ln;

  const u16* arow = attn_bf + (size_t)nh * 1024;
  BFrag a0, a1, b;
  const u16* ap0 = arow + (size_t)ln * 32 + hi * 8;
  a0.q[0] = *(const u32x4*)ap0;
  a0.q[1] = *(const u32x4*)(ap0 + 16);
  const u16* ap1 = arow + (size_t)(16 + ln) * 32 + hi * 8;
  a1.q[0] = *(const u32x4*)ap1;
  a1.q[1] = *(const u32x4*)(ap1 + 16);
  const u16* bp = vT + ((size_t)n * HW + p) * DIM + h * 32 + hi * 8;
  b.q[0] = *(const u32x4*)bp;
  b.q[1] = *(const u32x4*)(bp + 16);

  const f32x8 z = {0.f, 0.f, 0.f, 0.f, 0.f, 0.f, 0.f, 0.f};
  f32x8 d0 = wmma_bf16(a0, b, z);
  f32x8 d1 = wmma_bf16(a1, b, z);

  u16* orow = outT + ((size_t)n * HW + p) * DIM + h * 32;
  *(u32x4*)(orow + hi * 8)      = pack8_bf16(d0);   // 8 consecutive channels
  *(u32x4*)(orow + 16 + hi * 8) = pack8_bf16(d1);
}

// ------------------------- [5] 1x1 projection + bias -----------------------
// grid (1024 image-rows, 3 m-tiles); same wave tiling as conv; K=192 (6 steps)
__global__ void proj_kernel(const u16* __restrict__ outT,
                            const u16* __restrict__ pw,
                            const float* __restrict__ bias,
                            float* __restrict__ out) {
  const int lane = threadIdx.x & 31;
  const int wid  = threadIdx.x >> 5;
  const int hi   = lane >> 4;
  const int ln   = lane & 15;
  const int prow = blockIdx.x;
  const int n    = prow >> 7;
  const int y    = prow & 127;
  const int mbase = blockIdx.y * 64 + (wid >> 1) * 16;
  const int nbase = (wid & 1) * 64;

  const f32x8 z = {0.f, 0.f, 0.f, 0.f, 0.f, 0.f, 0.f, 0.f};
  f32x8 acc[4] = {z, z, z, z};

#pragma unroll 2
  for (int ks = 0; ks < 6; ++ks) {
    const int kb = ks * 32;
    BFrag a;
    const u16* ap = pw + (size_t)(mbase + ln) * DIM + kb + hi * 8;
    a.q[0] = *(const u32x4*)ap;
    a.q[1] = *(const u32x4*)(ap + 16);
#pragma unroll
    for (int sub = 0; sub < 4; ++sub) {
      const int p = y * 128 + nbase + sub * 16 + ln;
      BFrag b;
      const u16* bp = outT + ((size_t)n * HW + p) * DIM + kb + hi * 8;
      b.q[0] = *(const u32x4*)bp;
      b.q[1] = *(const u32x4*)(bp + 16);
      acc[sub] = wmma_bf16(a, b, acc[sub]);
    }
  }

#pragma unroll
  for (int sub = 0; sub < 4; ++sub) {
    const int p = y * 128 + nbase + sub * 16 + ln;
#pragma unroll
    for (int g = 0; g < 8; ++g) {
      const int m = mbase + g + hi * 8;
      out[((size_t)n * DIM + m) * HW + p] = acc[sub][g] + bias[m];
    }
  }
}

// ---------------------------------------------------------------------------
extern "C" void kernel_launch(void* const* d_in, const int* in_sizes, int n_in,
                              void* d_out, int out_size, void* d_ws, size_t ws_size,
                              hipStream_t stream) {
  (void)in_sizes; (void)n_in; (void)out_size; (void)ws_size;
  const float* x      = (const float*)d_in[0];  // [8,192,128,128]
  const float* qkv_w  = (const float*)d_in[1];  // [576,192,3,3]
  const float* proj_w = (const float*)d_in[2];  // [192,192,1,1]
  const float* proj_b = (const float*)d_in[3];  // [192]
  const float* temp   = (const float*)d_in[4];  // [6,1,1]
  float* out = (float*)d_out;

  const size_t XB  = (size_t)8 * DIM * HW;          // 25,165,824 elements
  const size_t XPB = (size_t)8 * PW * PW * DIM;     // padded: 25,958,400 elems
  unsigned char* ws = (unsigned char*)d_ws;
  size_t off = 0;
  auto alloc = [&](size_t bytes) {
    void* p = ws + off;
    off += (bytes + 255) & ~(size_t)255;
    return p;
  };
  u16*   xTp    = (u16*)alloc(XPB * 2);
  u16*   q_bf   = (u16*)alloc(XB * 2);
  u16*   k_bf   = (u16*)alloc(XB * 2);
  u16*   vT_bf  = (u16*)alloc(XB * 2);
  u16*   outT   = (u16*)alloc(XB * 2);
  u16*   wpack  = (u16*)alloc((size_t)576 * 1728 * 2);
  u16*   pw_bf  = (u16*)alloc((size_t)DIM * DIM * 2);
  float* invq   = (float*)alloc((size_t)48 * 32 * 4);
  float* invk   = (float*)alloc((size_t)48 * 32 * 4);
  u16*   attn   = (u16*)alloc((size_t)48 * 1024 * 2);

  const dim3 blk(256);
  // [0] zero the padded input, then conversions / packing
  const int n16 = (int)(XPB * 2 / 16);              // u32x4 elements
  zero_u32x4_kernel<<<dim3((n16 + 255) / 256), blk, 0, stream>>>((u32x4*)xTp, n16);
  cvt_x_transpose_kernel<<<dim3(2048), blk, 0, stream>>>(x, xTp);
  pack_qkvw_kernel<<<dim3((576 * 1728 + 255) / 256), blk, 0, stream>>>(qkv_w, wpack);
  cvt_f32_bf16_kernel<<<dim3((DIM * DIM + 255) / 256), blk, 0, stream>>>(proj_w, pw_bf, DIM * DIM);
  // [1] qkv conv
  qkv_conv_kernel<<<dim3(1024, 9), blk, 0, stream>>>(xTp, wpack, q_bf, k_bf, vT_bf);
  // [2] norms
  norm_kernel<<<dim3(48), blk, 0, stream>>>(q_bf, k_bf, invq, invk);
  // [3] attention + softmax
  attn_kernel<<<dim3(48), blk, 0, stream>>>(q_bf, k_bf, invq, invk, temp, attn);
  // [4] attn @ v
  av_kernel<<<dim3(128, 48), blk, 0, stream>>>(attn, vT_bf, outT);
  // [5] projection + bias
  proj_kernel<<<dim3(1024, 3), blk, 0, stream>>>(outT, pw_bf, proj_b, out);
}
